// CrissCrossAttention_42485816492492
// MI455X (gfx1250) — matrix operand order repaired and checked by reading, not compiled, not run on hardware
//
#include <hip/hip_runtime.h>

// Criss-Cross Attention for MI455X (gfx1250), bf16 WMMA + fp32 accum.
// All LDS operand tiles are stored "fragment-linear" so every WMMA fragment
// load is 2x ds_load_b128; weight/value tiles stage via async global->LDS
// (ASYNCcnt); transposing stages go through VGPRs with scatter stores.
//
// Dims: N=8, C=512, CQK=64, H=W=96, P=9216.
// Workspace layout (bytes, 256-aligned), total ~180 MB:
//   wall   : bf16 [640][512]        @ 0          (655,360)
//   ball   : f32  [640]             @ 655,360    (2,560)
//   z      : bf16 [N][640][P]       @ 657,920    (94,371,840)   (q|k|v fused)
//   energy : f32  [N][192][P]       @ 95,029,760 (56,623,104)
//   attn   : bf16 [N][192][P]       @ 151,652,864(28,311,552)

#define NB    8
#define CIN   512
#define CQKd  64
#define HH    96
#define WW    96
#define PP    (HH * WW)           // 9216
#define MALL  (CQKd + CQKd + CIN) // 640
#define NEGV  (-1e30f)

typedef __bf16 bf16;
typedef __attribute__((ext_vector_type(16))) __bf16 v16bf;
typedef __attribute__((ext_vector_type(8)))  __bf16 v8bf;
typedef __attribute__((ext_vector_type(8)))  float  v8f;

static __device__ __forceinline__ v8f wmma_bf16(v16bf a, v16bf b, v8f c) {
  // D = A(16x32) * B(32x16) + C, fp32 accum
  return __builtin_amdgcn_wmma_f32_16x16x32_bf16(false, a, false, b, (short)0, c,
                                                 false, false);
}

// Generic LDS pointer -> 32-bit LDS byte offset. Flat aperture places the LDS
// offset in addr[31:0] (ISA 10.2 per-aperture calc), so truncation is exact.
static __device__ __forceinline__ unsigned lds_off(const void* p) {
  return (unsigned)(unsigned long long)(uintptr_t)p;
}

// Async 16B copy global -> LDS, tracked by ASYNCcnt (ISA 10 / 15.18.3).
static __device__ __forceinline__ void async_b128(void* lds_dst, const void* gsrc) {
  unsigned d = lds_off(lds_dst);
  unsigned long long s = (unsigned long long)(uintptr_t)gsrc;
  asm volatile("global_load_async_to_lds_b128 %0, %1, off"
               :: "v"(d), "v"(s) : "memory");
}

static __device__ __forceinline__ void wait_async0() {
  asm volatile("s_wait_asynccnt 0x0" ::: "memory");
}

// A fragment, A stored row-major in LDS: A[m][k] = s[m*ld + k].
// lane&15 = M row; lanes<16 hold K 0-15, lanes>=16 hold K 16-31.
// K runs contiguous -> 2x ds_load_b128.
static __device__ __forceinline__ v16bf frag_a_row(const bf16* s, int ld, int m0,
                                                   int k0, int lane) {
  v16bf a;
  const int m  = m0 + (lane & 15);
  const int kb = k0 + ((lane >> 4) << 4);
#pragma unroll
  for (int e = 0; e < 16; ++e) {
    const int k = kb + e;
    a[e] = s[m * ld + k];
  }
  return a;
}

// B fragment, B stored column-major in LDS: B[k][n] = s[n*ld + k].
// Layout: lane&15 = N col; lanes<16 hold K 0-15, lanes>=16 hold K 16-31.
// Per-lane K is one contiguous 16-element run -> 2x ds_load_b128.
static __device__ __forceinline__ v16bf frag_b_cm(const bf16* s, int ld, int n0,
                                                  int k0, int lane) {
  v16bf b;
  const int n  = n0 + (lane & 15);
  const int kb = k0 + ((lane >> 4) << 4);
#pragma unroll
  for (int e = 0; e < 16; ++e) b[e] = s[n * ld + kb + e];
  return b;
}

// ---------------------------------------------------------------- prep -----
__global__ __launch_bounds__(256) void k_prep(
    const float* __restrict__ wq, const float* __restrict__ wk,
    const float* __restrict__ wv, const float* __restrict__ bq,
    const float* __restrict__ bk, const float* __restrict__ bv,
    bf16* __restrict__ wall, float* __restrict__ ball) {
  const int stride = gridDim.x * blockDim.x;
  int idx = blockIdx.x * blockDim.x + threadIdx.x;
  for (int i = idx; i < MALL * CIN; i += stride) {
    const int o = i / CIN, c = i % CIN;
    float w;
    if (o < CQKd)            w = wq[o * CIN + c];
    else if (o < 2 * CQKd)   w = wk[(o - CQKd) * CIN + c];
    else                     w = wv[(o - 2 * CQKd) * CIN + c];
    wall[i] = (bf16)w;
  }
  if (idx < MALL) {
    float b;
    if (idx < CQKd)          b = bq[idx];
    else if (idx < 2 * CQKd) b = bk[idx - CQKd];
    else                     b = bv[idx - 2 * CQKd];
    ball[idx] = b;
  }
}

// ------------------------------------------------------------ QKV GEMM -----
// z[n][o][p] = sum_c wall[o][c] * x[n][c][p] + ball[o], o in [0,640)
// Grid: (P/64, 640/64, N). 8 waves/WG; each wave owns a 32x16 accum pair.
// A (weights) staged async in natural layout; B (x) converted fp32->bf16 and
// stored transposed [p][c] so B fragments are contiguous.
__global__ __launch_bounds__(256) void k_qkv_gemm(
    const float* __restrict__ x, const bf16* __restrict__ wall,
    const float* __restrict__ ball, bf16* __restrict__ z) {
  constexpr int LDA = 40, LDB = 40;
  __shared__ __align__(16) bf16 sA[64 * LDA];   // wall tile [64 o][32 c]
  __shared__ __align__(16) bf16 sBt[64 * LDB];  // x^T tile  [64 p][32 c]
  const int tid = threadIdx.x, lane = tid & 31, wave = tid >> 5;
  const int p0 = blockIdx.x * 64;
  const int o0 = blockIdx.y * 64;
  const int nb = blockIdx.z;
  const int wm = (wave & 1) * 32;
  const int wn = (wave >> 1) * 16;
  v8f acc0 = {}, acc1 = {};
  for (int k0 = 0; k0 < CIN; k0 += 32) {
    {  // stage A: async global->LDS, 16B per lane, no VGPR round-trip
      const int row = tid >> 2, seg = tid & 3;
      async_b128(&sA[row * LDA + seg * 8],
                 wall + (size_t)(o0 + row) * CIN + k0 + seg * 8);
    }
    {  // stage B: fp32 -> bf16 convert + transpose ([c][p] -> [p][c])
      const int row = tid >> 3, seg = tid & 7;  // row = c, seg = p-octet
      const float4* g =
          (const float4*)(x + (size_t)(nb * CIN + k0 + row) * PP + p0 + seg * 8);
      const float4 f0 = g[0], f1 = g[1];
      bf16* d = &sBt[(seg * 8) * LDB + row];
      d[0 * LDB] = (bf16)f0.x; d[1 * LDB] = (bf16)f0.y;
      d[2 * LDB] = (bf16)f0.z; d[3 * LDB] = (bf16)f0.w;
      d[4 * LDB] = (bf16)f1.x; d[5 * LDB] = (bf16)f1.y;
      d[6 * LDB] = (bf16)f1.z; d[7 * LDB] = (bf16)f1.w;
    }
    wait_async0();
    __syncthreads();
    const v16bf bfr = frag_b_cm(sBt, LDB, wn, 0, lane);
    const v16bf a0  = frag_a_row(sA, LDA, wm, 0, lane);
    const v16bf a1  = frag_a_row(sA, LDA, wm + 16, 0, lane);
    acc0 = wmma_bf16(a0, bfr, acc0);
    acc1 = wmma_bf16(a1, bfr, acc1);
    __syncthreads();
  }
  const int cc = wn + (lane & 15);
  const int rb = ((lane >> 4) & 1) * 8;
#pragma unroll
  for (int r = 0; r < 8; ++r) {
    const int o = o0 + wm + rb + r;
    z[(size_t)(nb * MALL + o) * PP + p0 + cc]      = (bf16)(acc0[r] + ball[o]);
    z[(size_t)(nb * MALL + o + 16) * PP + p0 + cc] = (bf16)(acc1[r] + ball[o + 16]);
  }
}

// --------------------------------------------------------- row energies ----
// E[i][x] = sum_c k[c][y*W+i] * q[c][y*W+x], per (n,y). K=64.
// Both operands staged transposed (vector global load + scatter store) so
// all fragments are contiguous ds_load_b128.
__global__ __launch_bounds__(256) void k_energy_row(const bf16* __restrict__ z,
                                                    float* __restrict__ energy) {
  constexpr int LDT = 72;                        // 64 c + pad
  __shared__ __align__(16) bf16 skT[HH * LDT];   // k^T [96 i][64 c]
  __shared__ __align__(16) bf16 sqT[WW * LDT];   // q^T [96 x][64 c]
  const int tid = threadIdx.x, lane = tid & 31, wave = tid >> 5;
  const int y = blockIdx.x, nb = blockIdx.y;
  const bf16* qg = z + (size_t)(nb * MALL) * PP + y * WW;
  const bf16* kg = z + (size_t)(nb * MALL + CQKd) * PP + y * WW;
  for (int idx = tid; idx < CQKd * 12; idx += 256) {
    const int c = idx / 12, seg = idx % 12;      // 8 pixels per segment
    const v8bf q8 = *(const v8bf*)(qg + (size_t)c * PP + seg * 8);
    const v8bf k8 = *(const v8bf*)(kg + (size_t)c * PP + seg * 8);
#pragma unroll
    for (int j = 0; j < 8; ++j) {
      sqT[(seg * 8 + j) * LDT + c] = q8[j];
      skT[(seg * 8 + j) * LDT + c] = k8[j];
    }
  }
  __syncthreads();
  for (int t = wave; t < 36; t += 8) {
    const int m0 = (t / 6) * 16, n0 = (t % 6) * 16;
    v8f acc = {};
#pragma unroll
    for (int k0 = 0; k0 < CQKd; k0 += 32) {
      const v16bf a = frag_a_row(skT, LDT, m0, k0, lane);  // A[i][c]
      const v16bf b = frag_b_cm(sqT, LDT, n0, k0, lane);   // B[c][x] = sqT[x][c]
      acc = wmma_bf16(a, b, acc);
    }
    const int xx = n0 + (lane & 15);
    const int rb = ((lane >> 4) & 1) * 8;
#pragma unroll
    for (int r = 0; r < 8; ++r) {
      const int i = m0 + rb + r;
      energy[(size_t)(nb * 192 + i) * PP + y * WW + xx] = acc[r];
    }
  }
}

// --------------------------------------------------------- col energies ----
// E[i][y] = sum_c k[c][i*W+x] * q[c][y*W+x], per (n,x); diag -> NEG. K=64.
__global__ __launch_bounds__(256) void k_energy_col(const bf16* __restrict__ z,
                                                    float* __restrict__ energy) {
  constexpr int LDT = 72;                        // 64 c + pad
  __shared__ __align__(16) bf16 skT[HH * LDT];   // k^T [96 i][64 c]
  __shared__ __align__(16) bf16 sqT[HH * LDT];   // q^T [96 y][64 c]
  const int tid = threadIdx.x, lane = tid & 31, wave = tid >> 5;
  const int xx = blockIdx.x, nb = blockIdx.y;
  for (int idx = tid; idx < CQKd * HH; idx += 256) {
    const int c = idx / HH, yy = idx % HH;
    sqT[yy * LDT + c] = z[(size_t)(nb * MALL + c) * PP + yy * WW + xx];
    skT[yy * LDT + c] = z[(size_t)(nb * MALL + CQKd + c) * PP + yy * WW + xx];
  }
  __syncthreads();
  for (int t = wave; t < 36; t += 8) {
    const int m0 = (t / 6) * 16, n0 = (t % 6) * 16;
    v8f acc = {};
#pragma unroll
    for (int k0 = 0; k0 < CQKd; k0 += 32) {
      const v16bf a = frag_a_row(skT, LDT, m0, k0, lane);  // A[i][c]
      const v16bf b = frag_b_cm(sqT, LDT, n0, k0, lane);   // B[c][y] = sqT[y][c]
      acc = wmma_bf16(a, b, acc);
    }
    const int yy = n0 + (lane & 15);
    const int rb = ((lane >> 4) & 1) * 8;
#pragma unroll
    for (int r = 0; r < 8; ++r) {
      const int i = m0 + rb + r;
      const float v = (i == yy) ? NEGV : acc[r];
      energy[(size_t)(nb * 192 + 96 + i) * PP + yy * WW + xx] = v;
    }
  }
}

// -------------------------------------------------------------- softmax ----
__global__ __launch_bounds__(256) void k_softmax(const float* __restrict__ energy,
                                                 bf16* __restrict__ attn) {
  const int g = blockIdx.x * blockDim.x + threadIdx.x;
  if (g >= NB * PP) return;
  const int nb = g / PP, p = g % PP;
  const float* e = energy + (size_t)nb * 192 * PP + p;
  float mx = -3.4e38f;
  for (int i = 0; i < 192; ++i) mx = fmaxf(mx, e[(size_t)i * PP]);
  float sum = 0.f;
  for (int i = 0; i < 192; ++i) sum += __expf(e[(size_t)i * PP] - mx);
  const float inv = 1.0f / sum;
  bf16* a = attn + (size_t)nb * 192 * PP + p;
  for (int i = 0; i < 192; ++i)
    a[(size_t)i * PP] = (bf16)(__expf(e[(size_t)i * PP] - mx) * inv);
}

// ------------------------------------------------ aggregation, row part ----
// out[c][x] = gamma * sum_i v[c][y*W+i] * a_row[i][y*W+x] + x_in, per (n,y).
// v tile staged async (natural layout is A-fragment-linear); attention tile
// staged transposed so B fragments are contiguous.
__global__ __launch_bounds__(256) void k_agg_row(
    const bf16* __restrict__ z, const bf16* __restrict__ attn,
    const float* __restrict__ xin, const float* __restrict__ gamma,
    float* __restrict__ out) {
  constexpr int LD = 104;
  __shared__ __align__(16) bf16 sv[64 * LD];   // v tile [64 c][96 i]
  __shared__ __align__(16) bf16 saT[WW * LD];  // a^T    [96 x][96 i]
  const int tid = threadIdx.x, lane = tid & 31, wave = tid >> 5;
  const int cbase = blockIdx.x * 64, y = blockIdx.y, nb = blockIdx.z;
  const bf16* vg = z + (size_t)(nb * MALL + 2 * CQKd + cbase) * PP + y * WW;
  for (int idx = tid; idx < 64 * 12; idx += 256) {
    const int row = idx / 12, seg = idx % 12;
    async_b128(&sv[row * LD + seg * 8], vg + (size_t)row * PP + seg * 8);
  }
  const bf16* ag = attn + (size_t)(nb * 192) * PP + y * WW;
  for (int idx = tid; idx < WW * 12; idx += 256) {
    const int i = idx / 12, seg = idx % 12;
    const v8bf a8 = *(const v8bf*)(ag + (size_t)i * PP + seg * 8);
#pragma unroll
    for (int j = 0; j < 8; ++j) saT[(seg * 8 + j) * LD + i] = a8[j];
  }
  wait_async0();
  __syncthreads();
  const float g = *gamma;
  for (int t = wave; t < 24; t += 8) {
    const int m0 = (t / 6) * 16, n0 = (t % 6) * 16;
    v8f acc = {};
#pragma unroll
    for (int k0 = 0; k0 < WW; k0 += 32) {
      const v16bf a = frag_a_row(sv, LD, m0, k0, lane);  // A[c][i]
      const v16bf b = frag_b_cm(saT, LD, n0, k0, lane);  // B[i][x] = saT[x][i]
      acc = wmma_bf16(a, b, acc);
    }
    const int xx = n0 + (lane & 15);
    const int rb = ((lane >> 4) & 1) * 8;
#pragma unroll
    for (int r = 0; r < 8; ++r) {
      const int c = cbase + m0 + rb + r;
      const size_t idx = (size_t)(nb * CIN + c) * PP + y * WW + xx;
      out[idx] = g * acc[r] + xin[idx];
    }
  }
}

// ------------------------------------------------ aggregation, col part ----
// out[c][y] += gamma * sum_i v[c][i*W+x] * a_col[i][y*W+x], per (n,x).
__global__ __launch_bounds__(256) void k_agg_col(
    const bf16* __restrict__ z, const bf16* __restrict__ attn,
    const float* __restrict__ gamma, float* __restrict__ out) {
  constexpr int LD = 104;
  __shared__ __align__(16) bf16 sv[64 * LD];   // v tile [64 c][96 i]
  __shared__ __align__(16) bf16 saT[HH * LD];  // a^T    [96 y][96 i]
  const int tid = threadIdx.x, lane = tid & 31, wave = tid >> 5;
  const int cbase = blockIdx.x * 64, xx = blockIdx.y, nb = blockIdx.z;
  for (int idx = tid; idx < 64 * HH; idx += 256) {
    const int c = idx / HH, i = idx % HH;
    sv[c * LD + i] =
        z[(size_t)(nb * MALL + 2 * CQKd + cbase + c) * PP + i * WW + xx];
  }
  for (int idx = tid; idx < HH * HH; idx += 256) {
    const int i = idx / HH, yy = idx % HH;
    saT[yy * LD + i] = attn[(size_t)(nb * 192 + 96 + i) * PP + yy * WW + xx];
  }
  __syncthreads();
  const float g = *gamma;
  for (int t = wave; t < 24; t += 8) {
    const int m0 = (t / 6) * 16, n0 = (t % 6) * 16;
    v8f acc = {};
#pragma unroll
    for (int k0 = 0; k0 < HH; k0 += 32) {
      const v16bf a = frag_a_row(sv, LD, m0, k0, lane);  // A[c][i]
      const v16bf b = frag_b_cm(saT, LD, n0, k0, lane);  // B[i][y] = saT[y][i]
      acc = wmma_bf16(a, b, acc);
    }
    const int yy = n0 + (lane & 15);
    const int rb = ((lane >> 4) & 1) * 8;
#pragma unroll
    for (int r = 0; r < 8; ++r) {
      const int c = cbase + m0 + rb + r;
      const size_t idx = (size_t)(nb * CIN + c) * PP + yy * WW + xx;
      out[idx] += g * acc[r];
    }
  }
}

// ---------------------------------------------------------------- launch ----
extern "C" void kernel_launch(void* const* d_in, const int* in_sizes, int n_in,
                              void* d_out, int out_size, void* d_ws, size_t ws_size,
                              hipStream_t stream) {
  (void)in_sizes; (void)n_in; (void)out_size; (void)ws_size;
  const float* x     = (const float*)d_in[0];
  const float* wq    = (const float*)d_in[1];
  const float* bq    = (const float*)d_in[2];
  const float* wk    = (const float*)d_in[3];
  const float* bk    = (const float*)d_in[4];
  const float* wv    = (const float*)d_in[5];
  const float* bv    = (const float*)d_in[6];
  const float* gamma = (const float*)d_in[7];
  float* out = (float*)d_out;

  char* ws = (char*)d_ws;
  bf16*  wall   = (bf16*)(ws);
  float* ball   = (float*)(ws + 655360);
  bf16*  z      = (bf16*)(ws + 657920);
  float* energy = (float*)(ws + 95029760);
  bf16*  attn   = (bf16*)(ws + 151652864);  // total ws need: 179,964,416 B

  k_prep<<<1280, 256, 0, stream>>>(wq, wk, wv, bq, bk, bv, wall, ball);
  k_qkv_gemm<<<dim3(PP / 64, MALL / 64, NB), 256, 0, stream>>>(x, wall, ball, z);
  k_energy_row<<<dim3(HH, NB), 256, 0, stream>>>(z, energy);
  k_energy_col<<<dim3(WW, NB), 256, 0, stream>>>(z, energy);
  k_softmax<<<(NB * PP + 255) / 256, 256, 0, stream>>>(energy, attn);
  k_agg_row<<<dim3(CIN / 64, HH, NB), 256, 0, stream>>>(z, attn, x, gamma, out);
  k_agg_col<<<dim3(CIN / 64, WW, NB), 256, 0, stream>>>(z, attn, gamma, out);
}
